// SurnameClassifierRNN_7078106104270
// MI455X (gfx1250) — compile-verified, matrix operand order checked by reading, not run on hardware
//
#include <hip/hip_runtime.h>

// ---------------------------------------------------------------------------
// SurnameClassifierRNN for MI455X (gfx1250, wave32, WMMA bf16, async-to-LDS)
//   B=1024, S=128, V=32000, E=512, H=1024, C=512
// Per-step fused GEMM: pre = [emb(x_t) | h_prev] @ [W_ih^T ; W_hh^T], K=1536,
// tanh epilogue with fused length-gather; MLP head reuses the tile engine.
// Double-buffered LDS: chunk i+1 streams in via GLOBAL_LOAD_ASYNC_TO_LDS_B128
// (ASYNCcnt) while chunk i runs through v_wmma_f32_16x16x32_bf16.
// ---------------------------------------------------------------------------

#define BB 1024
#define SS 128
#define EE 512
#define HH 1024
#define CC 512

typedef __attribute__((ext_vector_type(16))) __bf16       v16bf;
typedef __attribute__((ext_vector_type(8)))  float        v8f;
typedef __attribute__((ext_vector_type(8)))  unsigned int v8u;

__device__ __forceinline__ unsigned short f2bf(float f) {
    unsigned int u = __builtin_bit_cast(unsigned int, f);
    u += 0x7FFFu + ((u >> 16) & 1u);            // round-to-nearest-even
    return (unsigned short)(u >> 16);
}
__device__ __forceinline__ unsigned int pack_bf2(float lo, float hi) {
    return (unsigned int)f2bf(lo) | ((unsigned int)f2bf(hi) << 16);
}

// LDS tile: [64 rows][16 u32 of k], row stride 20 u32 (80 B):
//  - 20*row mod 64 distinct for rows 0..15 -> conflict-free fragment reads
//  - every (row, quad) 16-B segment is 16-B aligned -> async B128 legal
#define LSTRIDE 20

// Async DMA: global -> LDS, 16 bytes per lane, tracked by ASYNCcnt.
// Generic pointers to __shared__ carry the LDS byte offset in their low 32b.
__device__ __forceinline__ void async_ld_b128(unsigned lds_off, const void* gptr) {
    asm volatile("global_load_async_to_lds_b128 %0, %1, off"
                 :: "v"(lds_off), "v"((unsigned long long)(uintptr_t)gptr)
                 : "memory");
}
__device__ __forceinline__ void wait_async0() {
    asm volatile("s_wait_asynccnt 0" ::: "memory");
}

// Fragment per CDNA5 ISA 16-bit A/B layout: lane r=lane&15, half=lane>>4;
// halves j: k = ((j>>3)<<4) + (j&7) + half*8 -> each fragment VGPR = one u32.
__device__ __forceinline__ v16bf load_frag(const unsigned int* p32, int rowBase, int lane) {
    int r    = lane & 15;
    int half = lane >> 4;
    int base = (rowBase + r) * LSTRIDE;
    v8u a;
#pragma unroll
    for (int i = 0; i < 4; ++i) a[i]     = p32[base +     half * 4 + i];
#pragma unroll
    for (int i = 0; i < 4; ++i) a[4 + i] = p32[base + 8 + half * 4 + i];
    return __builtin_bit_cast(v16bf, a);
}

// MODE 0: RNN step  A=[emb|h_prev], act=tanh, out bf16 h_next + y_sel scatter
// MODE 1: MLP1      A=y_sel bf16,   act=relu, out bf16 z
// MODE 2: MLP2      A=z bf16,       act=none, out f32
template <int MODE>
__global__ __launch_bounds__(256)
void rnn_wmma_gemm(int t, int N, int Ktot,
                   const int* __restrict__ x_in, const int* __restrict__ x_len,
                   const float* __restrict__ emb,
                   const unsigned short* __restrict__ A_bf,
                   const unsigned short* __restrict__ h_cur,
                   const unsigned short* __restrict__ Wih_bf,
                   const unsigned short* __restrict__ Whh_bf,
                   const unsigned short* __restrict__ B_bf,
                   const float* __restrict__ bias,
                   unsigned short* __restrict__ out_bf,
                   unsigned short* __restrict__ y_sel,
                   float* __restrict__ out_f32) {
    __shared__ unsigned int lds_a[2][64 * LSTRIDE];
    __shared__ unsigned int lds_b[2][64 * LSTRIDE];

    const int tid    = threadIdx.x;
    const int lane   = tid & 31;
    const int wid    = tid >> 5;          // 8 waves
    const int wave_m = wid & 3;           // 4 M sub-tiles of 16
    const int wave_n = wid >> 2;          // 2 N sub-tiles of 32
    const int m0     = blockIdx.x * 64;
    const int n0     = blockIdx.y * 64;

    // fill mapping: thread owns (row = tid>>2, 8 consecutive k at quad*8)
    const int frow = tid >> 2;            // 0..63
    const int fq   = tid & 3;             // 0..3
    const int ldw  = frow * LSTRIDE + fq * 4;

    const unsigned aoff0 = (unsigned)(uintptr_t)&lds_a[0][ldw];
    const unsigned aoff1 = (unsigned)(uintptr_t)&lds_a[1][ldw];
    const unsigned boff0 = (unsigned)(uintptr_t)&lds_b[0][ldw];
    const unsigned boff1 = (unsigned)(uintptr_t)&lds_b[1][ldw];

    v8f c0 = {};
    v8f c1 = {};

    int tok = 0;
    if (MODE == 0) tok = x_in[(m0 + frow) * SS + t];   // hoisted gather index

    // issue chunk c's tile fills into buffer `buf`
    auto fill = [&](int c, int buf) {
        const int k0 = c * 32;                         // uniform (scalar)
        const unsigned ao = buf ? aoff1 : aoff0;
        const unsigned bo = buf ? boff1 : boff0;
        if (MODE == 0) {
            const unsigned short* bsrc =
                (k0 < EE) ? Wih_bf + (n0 + frow) * EE + k0 + fq * 8
                          : Whh_bf + (n0 + frow) * HH + (k0 - EE) + fq * 8;
            async_ld_b128(bo, bsrc);
            if (k0 < EE) {
                const float4* ep = (const float4*)(emb + tok * EE + k0 + fq * 8);
                float4 f0 = ep[0], f1 = ep[1];
                unsigned int* d = &lds_a[buf][ldw];
                d[0] = pack_bf2(f0.x, f0.y);
                d[1] = pack_bf2(f0.z, f0.w);
                d[2] = pack_bf2(f1.x, f1.y);
                d[3] = pack_bf2(f1.z, f1.w);
            } else {
                async_ld_b128(ao, h_cur + (m0 + frow) * HH + (k0 - EE) + fq * 8);
            }
        } else {
            async_ld_b128(ao, A_bf + (m0 + frow) * Ktot + k0 + fq * 8);
            async_ld_b128(bo, B_bf + (n0 + frow) * Ktot + k0 + fq * 8);
        }
    };

    const int nch = Ktot / 32;
    fill(0, 0);
    wait_async0();
    __syncthreads();

    for (int c = 0; c < nch; ++c) {
        const int cur = c & 1;
        if (c + 1 < nch) fill(c + 1, cur ^ 1);         // overlap DMA with WMMA

        const unsigned int* pa = lds_a[cur];
        const unsigned int* pb = lds_b[cur];
        v16bf a  = load_frag(pa, wave_m * 16, lane);
        v16bf b0 = load_frag(pb, wave_n * 32, lane);
        v16bf b1 = load_frag(pb, wave_n * 32 + 16, lane);
        c0 = __builtin_amdgcn_wmma_f32_16x16x32_bf16(false, a, false, b0, (short)0, c0, false, false);
        c1 = __builtin_amdgcn_wmma_f32_16x16x32_bf16(false, a, false, b1, (short)0, c1, false, false);

        wait_async0();                                 // chunk c+1 landed in LDS
        __syncthreads();
    }

    // ---------------- epilogue: D lanes 0-15 -> M=r, 16-31 -> M=8+r --------
    const int half = lane >> 4;
    const int nloc = lane & 15;
#pragma unroll
    for (int r = 0; r < 8; ++r) {
        int row  = m0 + wave_m * 16 + half * 8 + r;
        int col0 = n0 + wave_n * 32 + nloc;
        int col1 = col0 + 16;
        float v0 = c0[r] + bias[col0];
        float v1 = c1[r] + bias[col1];
        if (MODE == 0) {
            v0 = tanhf(v0);
            v1 = tanhf(v1);
            unsigned short r0 = f2bf(v0), r1 = f2bf(v1);
            out_bf[row * N + col0] = r0;
            out_bf[row * N + col1] = r1;
            if (x_len[row] - 1 == t) {
                y_sel[row * N + col0] = r0;
                y_sel[row * N + col1] = r1;
            }
        } else if (MODE == 1) {
            v0 = v0 > 0.f ? v0 : 0.f;
            v1 = v1 > 0.f ? v1 : 0.f;
            out_bf[row * N + col0] = f2bf(v0);
            out_bf[row * N + col1] = f2bf(v1);
        } else {
            out_f32[row * N + col0] = v0;
            out_f32[row * N + col1] = v1;
        }
    }
}

// ---------------------------- prep kernels ---------------------------------
__global__ void f32_to_bf16_kernel(const float* __restrict__ src,
                                   unsigned short* __restrict__ dst, int n) {
    int i = blockIdx.x * blockDim.x + threadIdx.x;
    if (i < n) dst[i] = f2bf(src[i]);
}
__global__ void bias_combine_kernel(const float* __restrict__ a,
                                    const float* __restrict__ b,
                                    float* __restrict__ c, int n) {
    int i = blockIdx.x * blockDim.x + threadIdx.x;
    if (i < n) c[i] = a[i] + b[i];
}
__global__ void zero_u16_kernel(unsigned short* __restrict__ p, int n) {
    int i = blockIdx.x * blockDim.x + threadIdx.x;
    if (i < n) p[i] = 0;
}

// ---------------------------------------------------------------------------
extern "C" void kernel_launch(void* const* d_in, const int* in_sizes, int n_in,
                              void* d_out, int out_size, void* d_ws, size_t ws_size,
                              hipStream_t stream) {
    (void)in_sizes; (void)n_in; (void)out_size; (void)ws_size;
    const int*   x_in  = (const int*)  d_in[0];
    const int*   x_len = (const int*)  d_in[1];
    const float* emb   = (const float*)d_in[2];
    const float* W_ih  = (const float*)d_in[3];
    const float* W_hh  = (const float*)d_in[4];
    const float* b_ih  = (const float*)d_in[5];
    const float* b_hh  = (const float*)d_in[6];
    const float* W1    = (const float*)d_in[7];
    const float* b1    = (const float*)d_in[8];
    const float* W2    = (const float*)d_in[9];
    const float* b2    = (const float*)d_in[10];
    float* out = (float*)d_out;

    char* ws = (char*)d_ws;
    unsigned short* wih_bf = (unsigned short*)(ws);                 // H*E
    unsigned short* whh_bf = (unsigned short*)(ws + (1u  << 20));   // H*H
    unsigned short* w1_bf  = (unsigned short*)(ws + (3u  << 20));   // H*H
    unsigned short* w2_bf  = (unsigned short*)(ws + (5u  << 20));   // C*H
    float*          bc     = (float*)        (ws + (6u  << 20));    // H f32
    unsigned short* hbuf0  = (unsigned short*)(ws + (6u  << 20) + 8192); // B*H
    unsigned short* hbuf1  = (unsigned short*)(ws + (8u  << 20) + 8192);
    unsigned short* ysel   = (unsigned short*)(ws + (10u << 20) + 8192);
    unsigned short* zbuf   = (unsigned short*)(ws + (12u << 20) + 8192);

    f32_to_bf16_kernel<<<(HH * EE + 255) / 256, 256, 0, stream>>>(W_ih, wih_bf, HH * EE);
    f32_to_bf16_kernel<<<(HH * HH + 255) / 256, 256, 0, stream>>>(W_hh, whh_bf, HH * HH);
    f32_to_bf16_kernel<<<(HH * HH + 255) / 256, 256, 0, stream>>>(W1,   w1_bf,  HH * HH);
    f32_to_bf16_kernel<<<(CC * HH + 255) / 256, 256, 0, stream>>>(W2,   w2_bf,  CC * HH);
    bias_combine_kernel<<<(HH + 255) / 256, 256, 0, stream>>>(b_ih, b_hh, bc, HH);
    zero_u16_kernel<<<(BB * HH + 255) / 256, 256, 0, stream>>>(hbuf0, BB * HH);

    // ---- 128 sequential RNN steps -----------------------------------------
    dim3 gridStep(BB / 64, HH / 64);   // 16 x 16 = 256 blocks
    for (int t = 0; t < SS; ++t) {
        unsigned short* hin  = (t & 1) ? hbuf1 : hbuf0;
        unsigned short* hout = (t & 1) ? hbuf0 : hbuf1;
        rnn_wmma_gemm<0><<<gridStep, 256, 0, stream>>>(
            t, HH, EE + HH,
            x_in, x_len, emb,
            (const unsigned short*)nullptr, hin, wih_bf, whh_bf,
            (const unsigned short*)nullptr, bc,
            hout, ysel, (float*)nullptr);
    }

    // ---- MLP head ---------------------------------------------------------
    dim3 gridM1(BB / 64, HH / 64);     // 16 x 16
    rnn_wmma_gemm<1><<<gridM1, 256, 0, stream>>>(
        0, HH, HH,
        x_in, x_len, emb,
        ysel, (const unsigned short*)nullptr,
        (const unsigned short*)nullptr, (const unsigned short*)nullptr,
        w1_bf, b1, zbuf, (unsigned short*)nullptr, (float*)nullptr);

    dim3 gridM2(BB / 64, CC / 64);     // 16 x 8
    rnn_wmma_gemm<2><<<gridM2, 256, 0, stream>>>(
        0, CC, HH,
        x_in, x_len, emb,
        zbuf, (const unsigned short*)nullptr,
        (const unsigned short*)nullptr, (const unsigned short*)nullptr,
        w2_bf, b2, (unsigned short*)nullptr, (unsigned short*)nullptr, out);
}
